// HeteroGNN_26774826124028
// MI455X (gfx1250) — matrix operand order repaired and checked by reading, not compiled, not run on hardware
//
#include <hip/hip_runtime.h>
#include <hip/hip_bf16.h>

typedef __attribute__((ext_vector_type(16))) __bf16 v16bf;
typedef __attribute__((ext_vector_type(8)))  float  v8f;

#define HID 64

__device__ __forceinline__ unsigned short f2bf(float f) {
  unsigned u = __float_as_uint(f);
  unsigned r = 0x7FFFu + ((u >> 16) & 1u);   // round-to-nearest-even
  return (unsigned short)((u + r) >> 16);
}

union Frag {
  v16bf v;
  uint4 q[2];
};

// C[M x 64] = A[M x K](f32) @ W[K x 64](f32) + bias, via bf16 WMMA, f32 accumulate.
// Block = 256 threads (8 waves), 128x64 output tile. K in {64,128,256}, kbits = log2(K).
// LDS: A tile [128][K+8] bf16 (row major), W tile transposed [64][K+8] bf16.
__global__ void gemm_wmma_bf16(const float* __restrict__ A,
                               const float* __restrict__ W,
                               const float* __restrict__ bias,
                               float* __restrict__ C,
                               int M, int K, int kbits) {
  extern __shared__ __align__(16) unsigned short lds[];
  const int KP = K + 8;
  unsigned short* ldsA = lds;             // [128][KP]
  unsigned short* ldsB = lds + 128 * KP;  // [64][KP]  (W^T: [n][k])

  const int rowBase = blockIdx.x * 128;
  const int t = threadIdx.x;

  // Stage A tile (f32 -> bf16), coalesced over k
  const int elemsA = 128 << kbits;
  for (int idx = t; idx < elemsA; idx += 256) {
    int r = idx >> kbits;
    int k = idx & (K - 1);
    int gr = rowBase + r;
    float v = (gr < M) ? A[(size_t)gr * K + k] : 0.0f;
    ldsA[r * KP + k] = f2bf(v);
  }
  // Stage W transposed (f32 -> bf16)
  const int elemsB = K * HID;
  for (int idx = t; idx < elemsB; idx += 256) {
    int k = idx >> 6;
    int n = idx & 63;
    ldsB[n * KP + k] = f2bf(W[idx]);
  }
  __syncthreads();

  const int wave = t >> 5;     // wave owns 16-row strip (M block)
  const int lane = t & 31;
  const int half = lane >> 4;  // K-half selector per ISA A/B bf16 layout
  const int lrow = lane & 15;
  const int row  = wave * 16 + lrow;

  // C/D layout: lanes 0-15 -> M=0..7 in v[0..7]; lanes 16-31 -> M=8..15. N = lane&15.
  v8f acc[4];
  for (int nb = 0; nb < 4; ++nb) {
    float bv = bias[nb * 16 + lrow];
    acc[nb] = (v8f){bv, bv, bv, bv, bv, bv, bv, bv};
  }

  for (int kk = 0; kk < K; kk += 32) {
    // A frag: lanes 0-15 hold K {kk..kk+7, kk+16..kk+23}; lanes 16-31 offset by +8.
    Frag a;
    a.q[0] = *(const uint4*)&ldsA[row * KP + kk + half * 8];
    a.q[1] = *(const uint4*)&ldsA[row * KP + kk + 16 + half * 8];
#pragma unroll
    for (int nb = 0; nb < 4; ++nb) {
      // B frag: lane holds column n = nb*16+lrow, K {kk+half*8..+7, kk+16+half*8..+7} (matches A).
      Frag b;
      const uint4* pb = (const uint4*)&ldsB[(nb * 16 + lrow) * KP + kk + half * 8];
      b.q[0] = pb[0];
      b.q[1] = pb[2];
      acc[nb] = __builtin_amdgcn_wmma_f32_16x16x32_bf16(
          false, a.v, false, b.v, (short)0, acc[nb], false, false);
    }
  }

#pragma unroll
  for (int nb = 0; nb < 4; ++nb)
#pragma unroll
    for (int r = 0; r < 8; ++r) {
      int gr = rowBase + wave * 16 + half * 8 + r;
      if (gr < M) C[(size_t)gr * HID + nb * 16 + lrow] = acc[nb][r];
    }
}

// One wave per edge, lane handles 2 channels. AGG pre-initialized with skip projection.
// Working set per conv (K/Q/V/AGG = 4 x 25.6 MB) fits in 192 MB L2 -> L2-resident scatter.
__global__ void edge_gate_agg(const int* __restrict__ ei, int E,
                              const float* __restrict__ Kd,
                              const float* __restrict__ Qs,
                              const float* __restrict__ Vs,
                              float* __restrict__ AGG) {
  int wid = (int)((blockIdx.x * blockDim.x + threadIdx.x) >> 5);
  int lane = threadIdx.x & 31;
  if (wid >= E) return;
  int s = ei[wid];
  int d = ei[E + wid];
  const float2 k = *(const float2*)&Kd[(size_t)d * HID + lane * 2];
  const float2 q = *(const float2*)&Qs[(size_t)s * HID + lane * 2];
  const float2 v = *(const float2*)&Vs[(size_t)s * HID + lane * 2];
  float gx = v.x / (1.0f + __expf(-(k.x + q.x)));
  float gy = v.y / (1.0f + __expf(-(k.y + q.y)));
  float* p = &AGG[(size_t)d * HID + lane * 2];
  atomicAdd(p, gx);
  atomicAdd(p + 1, gy);
}

// HeteroConv aggr='max' across edge types + leaky_relu(0.01)
__global__ void combine_max_lrelu(const float* __restrict__ a,
                                  const float* __restrict__ b,
                                  float* __restrict__ o, int n) {
  int i = blockIdx.x * blockDim.x + threadIdx.x;
  if (i < n) {
    float x = fmaxf(a[i], b[i]);
    o[i] = x > 0.0f ? x : 0.01f * x;
  }
}

// out[N,2] = X[N,64] @ W[64,2] + b  (too skinny for WMMA; VALU dot)
__global__ void final_lin(const float* __restrict__ X, const float* __restrict__ W,
                          const float* __restrict__ b, float* __restrict__ out, int N) {
  int i = blockIdx.x * blockDim.x + threadIdx.x;
  if (i >= N) return;
  float a0 = b[0], a1 = b[1];
  const float* x = X + (size_t)i * HID;
#pragma unroll
  for (int h = 0; h < HID; ++h) {
    float xv = x[h];
    a0 = fmaf(xv, W[h * 2], a0);
    a1 = fmaf(xv, W[h * 2 + 1], a1);
  }
  out[i * 2] = a0;
  out[i * 2 + 1] = a1;
}

extern "C" void kernel_launch(void* const* d_in, const int* in_sizes, int n_in,
                              void* d_out, int out_size, void* d_ws, size_t ws_size,
                              hipStream_t stream) {
  const float* x_item = (const float*)d_in[0];   // [NI,256]
  const float* x_user = (const float*)d_in[1];   // [NU,128]
  const int* ei_iu = (const int*)d_in[2];
  const int* ei_ui = (const int*)d_in[3];
  const int* ei_ii = (const int*)d_in[4];
  const int* ei_uu = (const int*)d_in[5];
  const int NI = in_sizes[0] / 256;
  const int NU = in_sizes[1] / 128;
  const int E  = in_sizes[2] / 2;

  // Workspace layout (all f32, HID channels per node)
  float* ws = (float*)d_ws;
  size_t nmax = (size_t)(NI > NU ? NI : NU) * HID;
  float* Kbuf       = ws;
  float* Qbuf       = Kbuf + nmax;
  float* Vbuf       = Qbuf + nmax;
  float* agg_item_a = Vbuf + nmax;
  float* agg_item_b = agg_item_a + (size_t)NI * HID;
  float* agg_user_a = agg_item_b + (size_t)NI * HID;
  float* agg_user_b = agg_user_a + (size_t)NU * HID;
  float* h_item     = agg_user_b + (size_t)NU * HID;
  float* h_user     = h_item + (size_t)NI * HID;

  auto in = [&](int i) { return (const float*)d_in[i]; };

  auto gemm = [&](const float* X, int N, int K, int kbits, const float* Wp,
                  const float* bp, float* out) {
    int grid = (N + 127) / 128;
    size_t sh = (size_t)(128 + 64) * (K + 8) * sizeof(unsigned short);
    gemm_wmma_bf16<<<grid, 256, sh, stream>>>(X, Wp, bp, out, N, K, kbits);
  };

  // conv param block: key.W, key.b, query.W, query.b, value.W, value.b, skip.W, skip.b
  auto conv = [&](int pb, const float* Xs, int Ns, int Ks, int ksb,
                  const float* Xd, int Nd, int Kd, int kdb,
                  const int* ei, float* agg) {
    gemm(Xd, Nd, Kd, kdb, in(pb + 0), in(pb + 1), Kbuf);  // key(dst)
    gemm(Xs, Ns, Ks, ksb, in(pb + 2), in(pb + 3), Qbuf);  // query(src)
    gemm(Xs, Ns, Ks, ksb, in(pb + 4), in(pb + 5), Vbuf);  // value(src)
    gemm(Xd, Nd, Kd, kdb, in(pb + 6), in(pb + 7), agg);   // skip(dst) initializes AGG
    int eblocks = (E + 7) / 8;                             // 8 waves/block, 1 edge/wave
    edge_gate_agg<<<eblocks, 256, 0, stream>>>(ei, E, Kbuf, Qbuf, Vbuf, agg);
  };

  // ---- Layer 0 (params start at index 6; conv order iu, ui, ii, uu) ----
  conv(6 + 0 * 8, x_item, NI, 256, 8, x_user, NU, 128, 7, ei_iu, agg_user_a);
  conv(6 + 1 * 8, x_user, NU, 128, 7, x_item, NI, 256, 8, ei_ui, agg_item_a);
  conv(6 + 2 * 8, x_item, NI, 256, 8, x_item, NI, 256, 8, ei_ii, agg_item_b);
  conv(6 + 3 * 8, x_user, NU, 128, 7, x_user, NU, 128, 7, ei_uu, agg_user_b);
  combine_max_lrelu<<<(NI * HID + 255) / 256, 256, 0, stream>>>(agg_item_a, agg_item_b, h_item, NI * HID);
  combine_max_lrelu<<<(NU * HID + 255) / 256, 256, 0, stream>>>(agg_user_a, agg_user_b, h_user, NU * HID);

  // ---- Layer 1 (params start at index 38; all dims = 64) ----
  conv(38 + 0 * 8, h_item, NI, 64, 6, h_user, NU, 64, 6, ei_iu, agg_user_a);
  conv(38 + 1 * 8, h_user, NU, 64, 6, h_item, NI, 64, 6, ei_ui, agg_item_a);
  conv(38 + 2 * 8, h_item, NI, 64, 6, h_item, NI, 64, 6, ei_ii, agg_item_b);
  conv(38 + 3 * 8, h_user, NU, 64, 6, h_user, NU, 64, 6, ei_uu, agg_user_b);
  combine_max_lrelu<<<(NI * HID + 255) / 256, 256, 0, stream>>>(agg_item_a, agg_item_b, h_item, NI * HID);
  combine_max_lrelu<<<(NU * HID + 255) / 256, 256, 0, stream>>>(agg_user_a, agg_user_b, h_user, NU * HID);

  // ---- Final linear on item nodes (params 70, 71) ----
  final_lin<<<(NI + 255) / 256, 256, 0, stream>>>(h_item, in(70), in(71), (float*)d_out, NI);
}
